// LangModel_60266981098073
// MI455X (gfx1250) — matrix operand order repaired and checked by reading, not compiled, not run on hardware
//
#include <hip/hip_runtime.h>

// Model dims
#define L_   8
#define B_   4
#define S_   1024
#define D_   1024
#define H_   16
#define V_   256
#define FF_  4096
#define HD_  64
#define LN_EPS 1e-5f

typedef unsigned short u16;
typedef unsigned int   u32;
typedef __attribute__((ext_vector_type(16))) __bf16 v16bf;
typedef __attribute__((ext_vector_type(8)))  float  v8f;

__device__ __forceinline__ u16 f2bf(float f) {
    union { float f; u32 u; } c; c.f = f;
    u32 u = c.u;
    return (u16)((u + 0x7FFFu + ((u >> 16) & 1u)) >> 16);
}

__device__ __forceinline__ void wait_async0() {
#if __has_builtin(__builtin_amdgcn_s_wait_asynccnt)
    __builtin_amdgcn_s_wait_asynccnt(0);
#else
    asm volatile("s_wait_asynccnt 0x0" ::: "memory");
#endif
}

// Async copy 16B global -> LDS (CDNA5 GLOBAL_LOAD_ASYNC_TO_LDS_B128, ASYNCcnt-tracked)
__device__ __forceinline__ void async_g2l_b128(u32 lds_byte_addr, const void* gaddr) {
    asm volatile("global_load_async_to_lds_b128 %0, %1, off"
                 :: "v"(lds_byte_addr), "v"(gaddr) : "memory");
}

// ---------------- elementwise kernels ----------------

__global__ void cvt_bf16_kernel(const float* __restrict__ src, u16* __restrict__ dst, long n) {
    long i = (long)blockIdx.x * blockDim.x + threadIdx.x;
    long stride = (long)gridDim.x * blockDim.x;
    for (; i < n; i += stride) dst[i] = f2bf(src[i]);
}

__global__ void embed_kernel(const int* __restrict__ tok, const float* __restrict__ te,
                             const float* __restrict__ pe, float* __restrict__ x) {
    int row = blockIdx.x;              // b*S + s
    int s   = row & (S_ - 1);
    int t   = tok[row];
    const float* tep = te + (long)t * D_;
    const float* pep = pe + (long)s * D_;
    float* xp = x + (long)row * D_;
    for (int i = threadIdx.x; i < D_; i += blockDim.x) xp[i] = tep[i] + pep[i];
}

__global__ __launch_bounds__(256)
void layernorm_kernel(const float* __restrict__ x, u16* __restrict__ h,
                      const float* __restrict__ g, const float* __restrict__ b) {
    __shared__ float red[256];
    int row = blockIdx.x;
    const float* xp = x + (long)row * D_;
    int tid = threadIdx.x;
    float v[4];
    float s = 0.f;
#pragma unroll
    for (int j = 0; j < 4; j++) { v[j] = xp[tid + j * 256]; s += v[j]; }
    red[tid] = s; __syncthreads();
    for (int o = 128; o > 0; o >>= 1) { if (tid < o) red[tid] += red[tid + o]; __syncthreads(); }
    float mu = red[0] * (1.f / D_); __syncthreads();
    s = 0.f;
#pragma unroll
    for (int j = 0; j < 4; j++) { float d = v[j] - mu; s += d * d; }
    red[tid] = s; __syncthreads();
    for (int o = 128; o > 0; o >>= 1) { if (tid < o) red[tid] += red[tid + o]; __syncthreads(); }
    float rs = rsqrtf(red[0] * (1.f / D_) + LN_EPS);
    u16* hp = h + (long)row * D_;
#pragma unroll
    for (int j = 0; j < 4; j++) {
        int c = tid + j * 256;
        hp[c] = f2bf((v[j] - mu) * rs * g[c] + b[c]);
    }
}

// Causal softmax over one row of scores; writes bf16 probs in-place over the
// fp32 row (probs row r lives at bf16-stride 2*S within the fp32 buffer).
__global__ __launch_bounds__(256)
void softmax_kernel(float* __restrict__ scores) {
    __shared__ float red[256];
    long r  = blockIdx.x;               // (b*H+h)*S + q
    long bh = r >> 10;
    int  q  = (int)(r & (S_ - 1));
    float* base = scores + bh * (long)(S_ * S_) + (long)q * S_;
    int tid = threadIdx.x;
    float v[4];
    float mx = -3.4e38f;
#pragma unroll
    for (int j = 0; j < 4; j++) {
        int c = tid + j * 256;
        v[j] = (c <= q) ? base[c] : -3.4e38f;
        mx = fmaxf(mx, v[j]);
    }
    red[tid] = mx; __syncthreads();
    for (int o = 128; o > 0; o >>= 1) { if (tid < o) red[tid] = fmaxf(red[tid], red[tid + o]); __syncthreads(); }
    mx = red[0]; __syncthreads();
    float s = 0.f;
#pragma unroll
    for (int j = 0; j < 4; j++) {
        int c = tid + j * 256;
        v[j] = (c <= q) ? __expf(v[j] - mx) : 0.f;
        s += v[j];
    }
    red[tid] = s; __syncthreads();
    for (int o = 128; o > 0; o >>= 1) { if (tid < o) red[tid] += red[tid + o]; __syncthreads(); }
    float inv = 1.f / red[0];
    u16* outp = (u16*)base;             // all reads done before first __syncthreads()
#pragma unroll
    for (int j = 0; j < 4; j++) outp[tid + j * 256] = f2bf(v[j] * inv);
}

// ---------------- WMMA GEMM ----------------
// C[M,N] = alpha * A[M,K] @ B[K,N] (+ bias[N]) with epilogue:
//   EPI 0: Cb = bf16(acc)            EPI 1: Cb = bf16(relu(acc+bias))
//   EPI 2: Cf += acc + bias          EPI 3: Cf = alpha*acc (+bias)
// BT: B is stored transposed (B_logical[k][n] = Bg[n*ldb + k]).
// Batched via z: off = (z/zdiv)*s1 + (z%zdiv)*s2 per operand.
// A tile staged with GLOBAL_LOAD_ASYNC_TO_LDS_B128 (ASYNCcnt, latency hidden
// behind the B staging); B tile staged with vector loads + v_perm_b32 packing
// into a k-pair-packed, n-major LDS layout so every WMMA fragment (A and B)
// is exactly 2x ds_load_b128.
template<int BM, int BN, int BK, int WR, int WC, int TM, int TN, int EPI, bool BT>
__global__ __launch_bounds__(32 * WR * WC)
void gemm_wmma_kernel(const u16* __restrict__ A, long a1, long a2, int lda,
                      const u16* __restrict__ B, long b1, long b2, int ldb,
                      const float* __restrict__ bias,
                      float* __restrict__ Cf, u16* __restrict__ Cb,
                      long c1, long c2, int ldc,
                      int K, int zdiv, float alpha) {
    constexpr int NT = 32 * WR * WC;
    static_assert(BK == 64, "BK fixed at 64 (two WMMA k-steps per stage)");
    static_assert(BM * BK / NT == 32, "A staging: 2 units of 32B per thread");
    static_assert(WR * TM * 16 == BM && WC * TN * 16 == BN, "tile shape");
    static_assert(!BT || BN == 128, "BT staging assumes BN==128");

    __shared__ __align__(16) u16 Alds[BM][BK + 8];          // row-major, 144B pitch
    __shared__ __align__(16) u32 Blds[BN][BK / 2 + 4];      // n-major k-pairs, 144B pitch

    int tid = threadIdx.x;
    int z = blockIdx.z;
    long zq = z / zdiv, zr = z % zdiv;
    const u16* Ag = A + zq * a1 + zr * a2;
    const u16* Bg = B + zq * b1 + zr * b2;
    long coff = zq * c1 + zr * c2;

    int m0 = blockIdx.y * BM;
    int n0 = blockIdx.x * BN;

    int lane = tid & 31;
    int wid  = tid >> 5;
    int wm = wid / WC, wn = wid % WC;
    int hf = lane >> 4;        // half-wave (ISA A/B operand layout split)
    int lm = lane & 15;

    v8f acc[TM][TN];
#pragma unroll
    for (int i = 0; i < TM; i++)
#pragma unroll
        for (int j = 0; j < TN; j++)
            acc[i][j] = (v8f){0.f, 0.f, 0.f, 0.f, 0.f, 0.f, 0.f, 0.f};

    for (int k0 = 0; k0 < K; k0 += BK) {
        // ---- stage A tile: async global -> LDS (4x B128 per thread) ----
#pragma unroll
        for (int u = tid; u < BM * BK / 16; u += NT) {
            int ar   = u / (BK / 16);
            int aseg = (u % (BK / 16)) * 16;
            const u16* g = Ag + (long)(m0 + ar) * lda + k0 + aseg;
            u32 al = (u32)(size_t)&Alds[ar][aseg];
            async_g2l_b128(al,      g);
            async_g2l_b128(al + 16, g + 8);
        }
        // ---- stage B tile into k-pair-packed n-major layout ----
        if (!BT) {
            constexpr int TPP = BN / 8;            // threads per k-pair-row
#pragma unroll
            for (int pr = tid / TPP; pr < BK / 2; pr += NT / TPP) {
                int nn0 = (tid % TPP) * 8;
                const u16* g0 = Bg + (long)(k0 + 2 * pr) * ldb + (n0 + nn0);
                union { uint4 q; u32 w[4]; } ra, rb;
                ra.q = *(const uint4*)g0;
                rb.q = *(const uint4*)(g0 + ldb);
#pragma unroll
                for (int i = 0; i < 4; i++) {      // interleave rows k/k+1 via v_perm_b32
                    Blds[nn0 + 2 * i    ][pr] = __builtin_amdgcn_perm(rb.w[i], ra.w[i], 0x05040100u);
                    Blds[nn0 + 2 * i + 1][pr] = __builtin_amdgcn_perm(rb.w[i], ra.w[i], 0x07060302u);
                }
            }
        } else {
            int nn   = tid >> 1;
            int kseg = (tid & 1) * 32;
            const u16* g0 = Bg + (long)(n0 + nn) * ldb + k0 + kseg;
#pragma unroll
            for (int i = 0; i < 4; i++) {          // k-pairs already adjacent in memory
                uint4 r = *(const uint4*)(g0 + 8 * i);
                *(uint4*)&Blds[nn][kseg / 2 + 4 * i] = r;
            }
        }
        wait_async0();
        __syncthreads();

        union AB { uint4 q[2]; u32 u[8]; v16bf v; };
#pragma unroll
        for (int kk = 0; kk < BK; kk += 32) {
            AB af[TM];
#pragma unroll
            for (int tm = 0; tm < TM; tm++) {
                // A 16x32 bf16 layout: lanes 0-15 K={0..7,16..23}, lanes 16-31 +8
                const u16* rowp = &Alds[wm * TM * 16 + tm * 16 + lm][kk];
                af[tm].q[0] = *(const uint4*)(rowp + 8 * hf);
                af[tm].q[1] = *(const uint4*)(rowp + 16 + 8 * hf);
            }
            AB bfr[TN];
#pragma unroll
            for (int tn = 0; tn < TN; tn++) {
                // B 32x16 layout: lanes 0-15 K={0..7,16..23}, lanes 16-31 +8 (pairs packed)
                const u32* bp = &Blds[wn * TN * 16 + tn * 16 + lm][kk / 2 + 4 * hf];
                bfr[tn].q[0] = *(const uint4*)bp;
                bfr[tn].q[1] = *(const uint4*)(bp + 8);
            }
#pragma unroll
            for (int tm = 0; tm < TM; tm++)
#pragma unroll
                for (int tn = 0; tn < TN; tn++)
                    acc[tm][tn] = __builtin_amdgcn_wmma_f32_16x16x32_bf16(
                        false, af[tm].v, false, bfr[tn].v, (short)0, acc[tm][tn], false, false);
        }
        __syncthreads();
    }

    // epilogue — C/D f32 16x16 layout: VGPR r -> row r (+8 upper half-wave), col = lane&15
#pragma unroll
    for (int tm = 0; tm < TM; tm++) {
        int mb = m0 + wm * TM * 16 + tm * 16;
#pragma unroll
        for (int tn = 0; tn < TN; tn++) {
            int nb = n0 + wn * TN * 16 + tn * 16 + lm;
            float bv = bias ? bias[nb] : 0.f;
#pragma unroll
            for (int r = 0; r < 8; r++) {
                int row = mb + r + 8 * hf;
                long idx = coff + (long)row * ldc + nb;
                float vv = acc[tm][tn][r] * alpha + bv;
                if (EPI == 0)      Cb[idx] = f2bf(vv);
                else if (EPI == 1) Cb[idx] = f2bf(fmaxf(vv, 0.f));
                else if (EPI == 2) Cf[idx] += vv;
                else               Cf[idx] = vv;
            }
        }
    }
}

// ---------------- host ----------------

extern "C" void kernel_launch(void* const* d_in, const int* in_sizes, int n_in,
                              void* d_out, int out_size, void* d_ws, size_t ws_size,
                              hipStream_t stream) {
    (void)in_sizes; (void)n_in; (void)out_size; (void)ws_size;

    const int*   tokens  = (const int*)  d_in[0];
    const float* tok_emb = (const float*)d_in[1];
    const float* pos_emb = (const float*)d_in[2];
    const float* Wq      = (const float*)d_in[3];
    const float* Wk      = (const float*)d_in[4];
    const float* Wv      = (const float*)d_in[5];
    const float* Wo      = (const float*)d_in[6];
    const float* bo      = (const float*)d_in[7];
    const float* ln1_g   = (const float*)d_in[8];
    const float* ln1_b   = (const float*)d_in[9];
    const float* ln2_g   = (const float*)d_in[10];
    const float* ln2_b   = (const float*)d_in[11];
    const float* W1      = (const float*)d_in[12];
    const float* b1      = (const float*)d_in[13];
    const float* W2      = (const float*)d_in[14];
    const float* b2      = (const float*)d_in[15];
    const float* lnf_g   = (const float*)d_in[16];
    const float* lnf_b   = (const float*)d_in[17];
    const float* Wout    = (const float*)d_in[18];
    const float* bout    = (const float*)d_in[19];

    char* p = (char*)d_ws;
    auto take = [&](size_t bytes) -> char* {
        char* r = p;
        p += (bytes + 255) & ~(size_t)255;
        return r;
    };
    u16*   wq_bf   = (u16*)take((size_t)L_ * D_ * D_ * 2);
    u16*   wk_bf   = (u16*)take((size_t)L_ * D_ * D_ * 2);
    u16*   wv_bf   = (u16*)take((size_t)L_ * D_ * D_ * 2);
    u16*   wo_bf   = (u16*)take((size_t)L_ * D_ * D_ * 2);
    u16*   w1_bf   = (u16*)take((size_t)L_ * D_ * FF_ * 2);
    u16*   w2_bf   = (u16*)take((size_t)L_ * FF_ * D_ * 2);
    u16*   wout_bf = (u16*)take((size_t)D_ * V_ * 2);
    float* x       = (float*)take((size_t)B_ * S_ * D_ * 4);
    u16*   h       = (u16*)take((size_t)B_ * S_ * D_ * 2);
    u16*   qb      = (u16*)take((size_t)B_ * S_ * D_ * 2);
    u16*   kb      = (u16*)take((size_t)B_ * S_ * D_ * 2);
    u16*   vb      = (u16*)take((size_t)B_ * S_ * D_ * 2);
    u16*   attno   = (u16*)take((size_t)B_ * S_ * D_ * 2);
    u16*   mid     = (u16*)take((size_t)B_ * S_ * FF_ * 2);
    float* scores  = (float*)take((size_t)B_ * H_ * S_ * S_ * 4);  // probs reuse as bf16

    dim3 blk(256);

    // weight f32 -> bf16 (deterministic per call, bandwidth-trivial)
    cvt_bf16_kernel<<<2048, blk, 0, stream>>>(Wq,   wq_bf,   (long)L_ * D_ * D_);
    cvt_bf16_kernel<<<2048, blk, 0, stream>>>(Wk,   wk_bf,   (long)L_ * D_ * D_);
    cvt_bf16_kernel<<<2048, blk, 0, stream>>>(Wv,   wv_bf,   (long)L_ * D_ * D_);
    cvt_bf16_kernel<<<2048, blk, 0, stream>>>(Wo,   wo_bf,   (long)L_ * D_ * D_);
    cvt_bf16_kernel<<<2048, blk, 0, stream>>>(W1,   w1_bf,   (long)L_ * D_ * FF_);
    cvt_bf16_kernel<<<2048, blk, 0, stream>>>(W2,   w2_bf,   (long)L_ * FF_ * D_);
    cvt_bf16_kernel<<<2048, blk, 0, stream>>>(Wout, wout_bf, (long)D_ * V_);

    embed_kernel<<<B_ * S_, blk, 0, stream>>>(tokens, tok_emb, pos_emb, x);

    dim3 gDD(D_ / 128, (B_ * S_) / 128, 1);            // [4096,1024] GEMMs
    dim3 gSC(S_ / 128, S_ / 128, B_ * H_);             // scores per (b,h)
    dim3 gAV(1, S_ / 128, B_ * H_);                    // attn @ V per (b,h)
    dim3 gF1(FF_ / 128, (B_ * S_) / 128, 1);           // FFN up
    dim3 gLO(V_ / 128, (B_ * S_) / 128, 1);            // logits

    for (int l = 0; l < L_; l++) {
        // h = LN1(x) in bf16
        layernorm_kernel<<<B_ * S_, blk, 0, stream>>>(x, h, ln1_g + (long)l * D_, ln1_b + (long)l * D_);

        // q/k/v = h @ W{q,k,v}[l]  (bf16 out)
        gemm_wmma_kernel<128,128,64,4,2,2,4,0,false><<<gDD, blk, 0, stream>>>(
            h, 0, 0, D_, wq_bf + (long)l * D_ * D_, 0, 0, D_,
            nullptr, nullptr, qb, 0, 0, D_, D_, 1, 1.f);
        gemm_wmma_kernel<128,128,64,4,2,2,4,0,false><<<gDD, blk, 0, stream>>>(
            h, 0, 0, D_, wk_bf + (long)l * D_ * D_, 0, 0, D_,
            nullptr, nullptr, kb, 0, 0, D_, D_, 1, 1.f);
        gemm_wmma_kernel<128,128,64,4,2,2,4,0,false><<<gDD, blk, 0, stream>>>(
            h, 0, 0, D_, wv_bf + (long)l * D_ * D_, 0, 0, D_,
            nullptr, nullptr, vb, 0, 0, D_, D_, 1, 1.f);

        // scores[b,h] = (Q @ K^T) * hd^-0.5   (fp32 out; K=64 -> single stage)
        gemm_wmma_kernel<128,128,64,4,2,2,4,3,true><<<gSC, blk, 0, stream>>>(
            qb, (long)S_ * D_, HD_, D_,
            kb, (long)S_ * D_, HD_, D_,
            nullptr, scores, nullptr,
            (long)H_ * S_ * S_, (long)S_ * S_, S_,
            HD_, H_, 0.125f);

        // causal softmax, bf16 probs in-place
        softmax_kernel<<<B_ * H_ * S_, blk, 0, stream>>>(scores);

        // attno[b,:,h*64:] = P @ V   (bf16 out)
        gemm_wmma_kernel<128,64,64,4,2,2,2,0,false><<<gAV, blk, 0, stream>>>(
            (const u16*)scores, 2L * H_ * S_ * S_, 2L * S_ * S_, 2 * S_,
            vb, (long)S_ * D_, HD_, D_,
            nullptr, nullptr, attno,
            (long)S_ * D_, HD_, D_,
            S_, H_, 1.f);

        // x += attno @ Wo[l] + bo[l]
        gemm_wmma_kernel<128,128,64,4,2,2,4,2,false><<<gDD, blk, 0, stream>>>(
            attno, 0, 0, D_, wo_bf + (long)l * D_ * D_, 0, 0, D_,
            bo + (long)l * D_, x, nullptr, 0, 0, D_, D_, 1, 1.f);

        // h = LN2(x)
        layernorm_kernel<<<B_ * S_, blk, 0, stream>>>(x, h, ln2_g + (long)l * D_, ln2_b + (long)l * D_);

        // mid = relu(h @ W1[l] + b1[l])  (bf16 out)
        gemm_wmma_kernel<128,128,64,4,2,2,4,1,false><<<gF1, blk, 0, stream>>>(
            h, 0, 0, D_, w1_bf + (long)l * D_ * FF_, 0, 0, FF_,
            b1 + (long)l * FF_, nullptr, mid, 0, 0, FF_, D_, 1, 1.f);

        // x += mid @ W2[l] + b2[l]
        gemm_wmma_kernel<128,128,64,4,2,2,4,2,false><<<gDD, blk, 0, stream>>>(
            mid, 0, 0, FF_, w2_bf + (long)l * FF_ * D_, 0, 0, D_,
            b2 + (long)l * D_, x, nullptr, 0, 0, D_, FF_, 1, 1.f);
    }

    // logits = LN_f(x) @ Wout + bout  (fp32 out to d_out)
    layernorm_kernel<<<B_ * S_, blk, 0, stream>>>(x, h, lnf_g, lnf_b);
    gemm_wmma_kernel<128,128,64,4,2,2,4,3,false><<<gLO, blk, 0, stream>>>(
        h, 0, 0, D_, wout_bf, 0, 0, V_,
        bout, (float*)d_out, nullptr, 0, 0, V_, D_, 1, 1.f);
}
